// TokenDownLayer_82815559402204
// MI455X (gfx1250) — matrix operand order repaired and checked by reading, not compiled, not run on hardware
//
#include <hip/hip_runtime.h>
#include <hip/hip_bf16.h>

typedef __attribute__((ext_vector_type(16))) _Float16 v16h;
typedef __attribute__((ext_vector_type(8)))  float    v8f;

#define Bb    32
#define Nn    1568
#define N0_   3136
#define NT    784
#define CIN   256
#define COUT  512
#define Hh    56
#define Ww    56
#define HS    28
#define WS_   28
#define HW    (Hh * Ww)      // 3136
#define HWS   (HS * WS_)     // 784
#define MROWS (Bb * NT)      // 25088
#define EPS_AGG 1e-6f
#define EPS_BN  1e-5f

__device__ __forceinline__ int pix_idx(float lx, float ly, int Hp, int Wp) {
    lx = fminf(fmaxf(lx, -1.f), 1.f) * 0.5f + 0.5f;
    ly = fminf(fmaxf(ly, -1.f), 1.f) * 0.5f + 0.5f;
    int ix = (int)rintf(lx * (float)(Wp - 1));   // round-half-even matches jnp.round
    int iy = (int)rintf(ly * (float)(Hp - 1));
    return iy * Wp + ix;
}

// ---- K1: token2map scatter: sums[B*HW*CIN], cnt[B*HW] ----
__global__ __launch_bounds__(256) void k_t2m(
    const float* __restrict__ x, const float* __restrict__ loc,
    const int* __restrict__ idx_agg,
    float* __restrict__ mapsum, float* __restrict__ mapcnt) {
    int gid  = blockIdx.x * 256 + threadIdx.x;       // B*N0*64 items
    int pair = gid >> 6;
    int lane = gid & 63;
    int b  = pair / N0_;
    int n0 = pair - b * N0_;
    int p  = pix_idx(loc[((size_t)b * N0_ + n0) * 2 + 0],
                     loc[((size_t)b * N0_ + n0) * 2 + 1], Hh, Ww);
    int tok = idx_agg[(size_t)b * N0_ + n0];
    const float* src = x + ((size_t)b * Nn + tok) * CIN + lane * 4;
    float* dst = mapsum + ((size_t)b * HW + p) * CIN + lane * 4;
    #pragma unroll
    for (int j = 0; j < 4; ++j) atomicAdd(dst + j, src[j]);
    if (lane == 0) atomicAdd(mapcnt + (size_t)b * HW + p, 1.0f);
}

// ---- K2: depthwise 3x3 stride2 conv on normalized map -> conv[B*HWS*CIN] ----
__global__ __launch_bounds__(256) void k_dwconv(
    const float* __restrict__ mapsum, const float* __restrict__ mapcnt,
    const float* __restrict__ dw_w, float* __restrict__ conv) {
    int e = blockIdx.x * 256 + threadIdx.x;          // B*HWS*CIN items
    int c = e & (CIN - 1);
    int p = (e >> 8) % HWS;
    int b = e / (CIN * HWS);
    int ho = p / WS_, wo = p - ho * WS_;
    float acc = 0.f;
    #pragma unroll
    for (int kh = 0; kh < 3; ++kh) {
        int hi = ho * 2 - 1 + kh;
        if (hi < 0 || hi >= Hh) continue;
        #pragma unroll
        for (int kw = 0; kw < 3; ++kw) {
            int wi = wo * 2 - 1 + kw;
            if (wi < 0 || wi >= Ww) continue;
            size_t pp = (size_t)b * HW + hi * Ww + wi;
            float v = mapsum[pp * CIN + c] / (mapcnt[pp] + EPS_AGG);
            acc += v * dw_w[c * 9 + kh * 3 + kw];
        }
    }
    conv[((size_t)b * HWS + p) * CIN + c] = acc;
}

// ---- K3: fused map2token + token_downup scatter ----
__global__ __launch_bounds__(256) void k_scatter2(
    const float* __restrict__ x, const float* __restrict__ conv,
    const float* __restrict__ loc, const int* __restrict__ idx_agg,
    const int* __restrict__ idx_agg_t, const float* __restrict__ agg_w_t,
    float* __restrict__ num_back, float* __restrict__ num_skip,
    float* __restrict__ den_t) {
    int gid  = blockIdx.x * 256 + threadIdx.x;
    int pair = gid >> 6;
    int lane = gid & 63;
    int b  = pair / N0_;
    int n0 = pair - b * N0_;
    int p  = pix_idx(loc[((size_t)b * N0_ + n0) * 2 + 0],
                     loc[((size_t)b * N0_ + n0) * 2 + 1], HS, WS_);
    int t  = idx_agg_t[(size_t)b * N0_ + n0];
    int s  = idx_agg[(size_t)b * N0_ + n0];
    float w = agg_w_t[(size_t)b * N0_ + n0];
    const float* g = conv + ((size_t)b * HWS + p) * CIN + lane * 4;
    const float* xs = x + ((size_t)b * Nn + s) * CIN + lane * 4;
    float* nb = num_back + ((size_t)b * NT + t) * CIN + lane * 4;
    float* ns = num_skip + ((size_t)b * NT + t) * CIN + lane * 4;
    #pragma unroll
    for (int j = 0; j < 4; ++j) {
        atomicAdd(nb + j, g[j] * w);
        atomicAdd(ns + j, xs[j] * w);
    }
    if (lane == 0) atomicAdd(den_t + (size_t)b * NT + t, w);
}

// ---- K4: y = skip*skip_w + back, fused BN1 partial stats ----
__global__ __launch_bounds__(256) void k_y_bn1(
    const float* __restrict__ num_back, const float* __restrict__ num_skip,
    const float* __restrict__ den_t, const float* __restrict__ skip_w,
    float* __restrict__ ybuf, float* __restrict__ bn1_sum, float* __restrict__ bn1_sq) {
    int c = threadIdx.x;                  // channel
    float sw = skip_w[c];
    float s = 0.f, ss = 0.f;
    int row0 = blockIdx.x * 32;
    for (int r = 0; r < 32; ++r) {
        int row = row0 + r;
        float inv = 1.0f / (den_t[row] + EPS_AGG);
        size_t o = (size_t)row * CIN + c;
        float y = num_skip[o] * inv * sw + num_back[o] * inv;
        ybuf[o] = y;
        s += y; ss += y * y;
    }
    atomicAdd(bn1_sum + c, s);
    atomicAdd(bn1_sq + c, ss);
}

// ---- K5/K8: BN finalize -> scale/shift ----
__global__ void k_bn_fin(const float* __restrict__ sum, const float* __restrict__ sq,
                         const float* __restrict__ gamma, const float* __restrict__ beta,
                         float* __restrict__ scale, float* __restrict__ shift, float invM) {
    int c = threadIdx.x + blockIdx.x * blockDim.x;
    float mu  = sum[c] * invM;
    float var = sq[c] * invM - mu * mu;
    float sc  = gamma[c] * rsqrtf(var + EPS_BN);
    scale[c] = sc;
    shift[c] = beta[c] - mu * sc;
}

// ---- K6: WMMA GEMM  z[M,COUT] = bn1(y)[M,CIN] @ conv_w[COUT,CIN]^T ----
__global__ __launch_bounds__(256) void k_gemm_wmma(
    const float* __restrict__ ybuf, const float* __restrict__ convw,
    const float* __restrict__ scale1, const float* __restrict__ shift1,
    float* __restrict__ zbuf) {
    __shared__ _Float16 As[32][40];   // [m][k], padded
    __shared__ _Float16 Bs[64][40];   // [n][k], padded
    const int tid  = threadIdx.x;
    const int lane = tid & 31;
    const int wid  = tid >> 5;
    const int waveM = wid & 1;        // 2 x 4 wave grid over 32x64 block tile
    const int waveN = wid >> 1;
    const int rowBlk = blockIdx.x * 32;
    const int colBlk = blockIdx.y * 64;

    v8f acc = {};
    for (int kc = 0; kc < 8; ++kc) {
        const int k0 = kc * 32;
        {   // stage A: 32x32, apply BN1 + cvt f16; 4 halves/thread
            int idx = tid * 4;
            int ar = idx >> 5, ac = idx & 31;
            const float* src = ybuf + ((size_t)(rowBlk + ar)) * CIN + k0 + ac;
            #pragma unroll
            for (int j = 0; j < 4; ++j) {
                int k = k0 + ac + j;
                As[ar][ac + j] = (_Float16)(src[j] * scale1[k] + shift1[k]);
            }
        }
        {   // stage B: 64x32 from conv_w[n][k] (k contiguous); 8 halves/thread
            int idx = tid * 8;
            int br = idx >> 5, bc = idx & 31;
            const float* src = convw + ((size_t)(colBlk + br)) * CIN + k0 + bc;
            #pragma unroll
            for (int j = 0; j < 8; ++j) Bs[br][bc + j] = (_Float16)src[j];
        }
        __syncthreads();
        // fragments per ISA 7.12.2 16-bit layouts (wave32)
        const bool hi = lane >= 16;
        const int mloc = waveM * 16 + (lane & 15);
        const int nloc = waveN * 16 + (lane & 15);
        v16h a, b;
        #pragma unroll
        for (int h = 0; h < 8; ++h) a[h]     = As[mloc][(hi ? 8 : 0)  + h];
        #pragma unroll
        for (int h = 0; h < 8; ++h) a[8 + h] = As[mloc][(hi ? 24 : 16) + h];
        #pragma unroll
        for (int h = 0; h < 8; ++h) b[h]     = Bs[nloc][(hi ? 8 : 0)  + h];
        #pragma unroll
        for (int h = 0; h < 8; ++h) b[8 + h] = Bs[nloc][(hi ? 24 : 16) + h];
        acc = __builtin_amdgcn_wmma_f32_16x16x32_f16(false, a, false, b,
                                                     (short)0, acc, false, false);
        __syncthreads();
    }
    const bool hi = lane >= 16;
    const int n = colBlk + waveN * 16 + (lane & 15);
    #pragma unroll
    for (int r = 0; r < 8; ++r) {
        int m = rowBlk + waveM * 16 + r + (hi ? 8 : 0);
        zbuf[(size_t)m * COUT + n] = acc[r];
    }
}

// ---- K7: BN2 partial stats over z ----
__global__ __launch_bounds__(256) void k_bn2_stats(
    const float* __restrict__ zbuf, float* __restrict__ bn2_sum, float* __restrict__ bn2_sq) {
    int c1 = threadIdx.x, c2 = threadIdx.x + 256;
    float s1 = 0.f, q1 = 0.f, s2 = 0.f, q2 = 0.f;
    int row0 = blockIdx.x * 32;
    for (int r = 0; r < 32; ++r) {
        size_t o = (size_t)(row0 + r) * COUT;
        float a = zbuf[o + c1], b = zbuf[o + c2];
        s1 += a; q1 += a * a; s2 += b; q2 += b * b;
    }
    atomicAdd(bn2_sum + c1, s1); atomicAdd(bn2_sq + c1, q1);
    atomicAdd(bn2_sum + c2, s2); atomicAdd(bn2_sq + c2, q2);
}

// ---- K9: BN2 + ReLU -> out ----
__global__ __launch_bounds__(256) void k_final(
    const float* __restrict__ zbuf, const float* __restrict__ scale2,
    const float* __restrict__ shift2, float* __restrict__ out) {
    size_t i = (size_t)blockIdx.x * 256 + threadIdx.x;   // MROWS*COUT items
    int c = (int)(i & (COUT - 1));
    out[i] = fmaxf(zbuf[i] * scale2[c] + shift2[c], 0.0f);
}

extern "C" void kernel_launch(void* const* d_in, const int* in_sizes, int n_in,
                              void* d_out, int out_size, void* d_ws, size_t ws_size,
                              hipStream_t stream) {
    const float* x         = (const float*)d_in[0];
    const float* loc_orig  = (const float*)d_in[1];
    const int*   idx_agg   = (const int*)  d_in[2];
    const int*   idx_agg_t = (const int*)  d_in[4];
    const float* agg_w_t   = (const float*)d_in[5];
    const float* dw_w      = (const float*)d_in[9];
    const float* skip_w    = (const float*)d_in[10];
    const float* conv_w    = (const float*)d_in[11];
    const float* gamma1    = (const float*)d_in[12];
    const float* beta1     = (const float*)d_in[13];
    const float* gamma2    = (const float*)d_in[14];
    const float* beta2     = (const float*)d_in[15];
    float* out = (float*)d_out;

    // workspace layout (floats); zeroed region first
    float* ws = (float*)d_ws;
    size_t o = 0;
    float* mapsum   = ws + o; o += (size_t)Bb * HW * CIN;   // 25,690,112
    float* mapcnt   = ws + o; o += (size_t)Bb * HW;
    float* num_back = ws + o; o += (size_t)MROWS * CIN;
    float* num_skip = ws + o; o += (size_t)MROWS * CIN;
    float* den_t    = ws + o; o += (size_t)MROWS;
    float* bn1_sum  = ws + o; o += CIN;
    float* bn1_sq   = ws + o; o += CIN;
    float* bn2_sum  = ws + o; o += COUT;
    float* bn2_sq   = ws + o; o += COUT;
    size_t zeroFloats = o;
    float* conv   = ws + o; o += (size_t)Bb * HWS * CIN;
    float* ybuf   = ws + o; o += (size_t)MROWS * CIN;
    float* zbuf   = ws + o; o += (size_t)MROWS * COUT;
    float* scale1 = ws + o; o += CIN;
    float* shift1 = ws + o; o += CIN;
    float* scale2 = ws + o; o += COUT;
    float* shift2 = ws + o; o += COUT;

    hipMemsetAsync(d_ws, 0, zeroFloats * sizeof(float), stream);

    const int scatterBlocks = (Bb * N0_ * 64) / 256;        // 25088
    k_t2m<<<scatterBlocks, 256, 0, stream>>>(x, loc_orig, idx_agg, mapsum, mapcnt);

    const int convBlocks = (Bb * HWS * CIN) / 256;          // 25088
    k_dwconv<<<convBlocks, 256, 0, stream>>>(mapsum, mapcnt, dw_w, conv);

    k_scatter2<<<scatterBlocks, 256, 0, stream>>>(x, conv, loc_orig, idx_agg,
                                                  idx_agg_t, agg_w_t,
                                                  num_back, num_skip, den_t);

    k_y_bn1<<<MROWS / 32, 256, 0, stream>>>(num_back, num_skip, den_t, skip_w,
                                            ybuf, bn1_sum, bn1_sq);

    k_bn_fin<<<1, CIN, 0, stream>>>(bn1_sum, bn1_sq, gamma1, beta1,
                                    scale1, shift1, 1.0f / (float)MROWS);

    dim3 gg(MROWS / 32, COUT / 64);                          // 784 x 8
    k_gemm_wmma<<<gg, 256, 0, stream>>>(ybuf, conv_w, scale1, shift1, zbuf);

    k_bn2_stats<<<MROWS / 32, 256, 0, stream>>>(zbuf, bn2_sum, bn2_sq);

    k_bn_fin<<<1, COUT, 0, stream>>>(bn2_sum, bn2_sq, gamma2, beta2,
                                     scale2, shift2, 1.0f / (float)MROWS);

    k_final<<<(int)(((size_t)MROWS * COUT) / 256), 256, 0, stream>>>(zbuf, scale2,
                                                                     shift2, out);
}
